// TemplatePairStackBlock_57561151701402
// MI455X (gfx1250) — matrix operand order repaired and checked by reading, not compiled, not run on hardware
//
#include <hip/hip_runtime.h>
#include <hip/hip_bf16.h>

// MI455X / gfx1250, wave32. All matrix contractions use v_wmma_f32_16x16x32_bf16
// (bf16 inputs, f32 accumulate). Working set (~34MB state + ~86MB ws streams) is
// L2-resident (192MB), so the kernel set is matrix-pipe bound, not HBM bound.

typedef __bf16 bf16;
typedef __attribute__((ext_vector_type(16))) __bf16 v16bf;
typedef __attribute__((ext_vector_type(8)))  float  v8f;

#define T_   2
#define N_   256
#define C_   64
#define H_   4
#define INFV 1e9f

// ---------------- WMMA fragment helpers (CDNA5 ISA §7.12.2 layouts) ----------

// A fragment: 16x32 bf16 tile. lane half 0: K 0-7 (V0-3), 16-23 (V4-7);
// half 1: K 8-15, 24-31. m = lane%16.
template <class F>
__device__ inline v16bf fragA(int lane, F f) {
  int m = lane & 15, half = (lane >> 4) & 1;
  v16bf r;
#pragma unroll
  for (int v = 0; v < 8; ++v) {
    int k0 = (v < 4 ? 2 * v : 2 * v + 8) + (half ? 8 : 0);
    r[2 * v]     = f(m, k0);
    r[2 * v + 1] = f(m, k0 + 1);
  }
  return r;
}

// B fragment: 32x16 bf16 tile. col n = lane%16; lane half 0: K 0-7, 16-23;
// half 1: K 8-15, 24-31 (same K order as fragA).
template <class F>
__device__ inline v16bf fragB(int lane, F f) {
  int n = lane & 15, half = (lane >> 4) & 1;
  v16bf r;
#pragma unroll
  for (int j = 0; j < 16; ++j) r[j] = f((j < 8 ? j : j + 8) + half * 8, n);
  return r;
}

// D (f32 16x16): VGPR r holds row m=r (lanes 0-15) / m=r+8 (lanes 16-31), col n=lane%16.
template <class F>
__device__ inline void storeD(int lane, v8f acc, F f) {
  int n = lane & 15, half = (lane >> 4) & 1;
#pragma unroll
  for (int r = 0; r < 8; ++r) f(r + half * 8, n, acc[r]);
}

__device__ inline v8f wmma_bf16(v16bf a, v16bf b, v8f c) {
  return __builtin_amdgcn_wmma_f32_16x16x32_bf16(false, a, false, b, (short)0, c,
                                                 false, false);
}

__device__ inline float sigmoidf_(float x) { return 1.f / (1.f + __expf(-x)); }

// ---------------- elementwise / LN kernels ----------------------------------

__global__ __launch_bounds__(256) void k_copy(const float* __restrict__ in,
                                              float* __restrict__ out, size_t n) {
  size_t i = blockIdx.x * (size_t)blockDim.x + threadIdx.x;
  size_t s = (size_t)gridDim.x * blockDim.x;
  for (; i < n; i += s) out[i] = in[i];
}

// LayerNorm over C=64 (row-major [rows,64]) -> bf16. 32 lanes per row.
__global__ __launch_bounds__(256) void k_ln(const float* __restrict__ x,
                                            const float* __restrict__ g,
                                            const float* __restrict__ b,
                                            bf16* __restrict__ out, int rows) {
  int row = blockIdx.x * (blockDim.x / 32) + (threadIdx.x / 32);
  if (row >= rows) return;
  int lane = threadIdx.x & 31;
  const float* xr = x + (size_t)row * C_;
  float v0 = xr[lane], v1 = xr[lane + 32];
  float s = v0 + v1, s2 = v0 * v0 + v1 * v1;
#pragma unroll
  for (int o = 16; o > 0; o >>= 1) {
    s += __shfl_xor(s, o, 32);
    s2 += __shfl_xor(s2, o, 32);
  }
  float mu = s * (1.f / 64.f);
  float var = s2 * (1.f / 64.f) - mu * mu;
  float inv = rsqrtf(var + 1e-5f);
  bf16* orow = out + (size_t)row * C_;
  orow[lane]      = (bf16)((v0 - mu) * inv * g[lane] + b[lane]);
  orow[lane + 32] = (bf16)((v1 - mu) * inv * g[lane + 32] + b[lane + 32]);
}

// LayerNorm over channel for channel-major x [T][64][N*N] -> bf16 row-major [T*N*N,64]
__global__ __launch_bounds__(256) void k_ln_cm(const float* __restrict__ x,
                                               const float* __restrict__ g,
                                               const float* __restrict__ b,
                                               bf16* __restrict__ out) {
  const int NN = N_ * N_;
  int row = blockIdx.x * (blockDim.x / 32) + (threadIdx.x / 32);
  if (row >= T_ * NN) return;
  int lane = threadIdx.x & 31;
  int t = row / NN, ij = row % NN;
  const float* base = x + (size_t)t * 64 * NN + ij;
  float v0 = base[(size_t)lane * NN], v1 = base[(size_t)(lane + 32) * NN];
  float s = v0 + v1, s2 = v0 * v0 + v1 * v1;
#pragma unroll
  for (int o = 16; o > 0; o >>= 1) {
    s += __shfl_xor(s, o, 32);
    s2 += __shfl_xor(s2, o, 32);
  }
  float mu = s * (1.f / 64.f);
  float var = s2 * (1.f / 64.f) - mu * mu;
  float inv = rsqrtf(var + 1e-5f);
  bf16* orow = out + (size_t)row * C_;
  orow[lane]      = (bf16)((v0 - mu) * inv * g[lane] + b[lane]);
  orow[lane + 32] = (bf16)((v1 - mu) * inv * g[lane + 32] + b[lane + 32]);
}

// tri_bias[t][h][a][b] = sum_c zn[t,a,b,c]*wb[c,h]
__global__ __launch_bounds__(256) void k_tribias(const bf16* __restrict__ zn,
                                                 const float* __restrict__ wb,
                                                 float* __restrict__ bias, int total) {
  int idx = blockIdx.x * blockDim.x + threadIdx.x;
  if (idx >= total) return;
  const bf16* zr = zn + (size_t)idx * C_;
  float a0 = 0.f, a1 = 0.f, a2 = 0.f, a3 = 0.f;
#pragma unroll 8
  for (int c = 0; c < C_; ++c) {
    float zv = (float)zr[c];
    a0 += zv * wb[c * H_ + 0];
    a1 += zv * wb[c * H_ + 1];
    a2 += zv * wb[c * H_ + 2];
    a3 += zv * wb[c * H_ + 3];
  }
  int t = idx / (N_ * N_), ab = idx % (N_ * N_);
  float* bb = bias + ((size_t)t * H_) * N_ * N_ + ab;
  bb[0 * N_ * N_] = a0;
  bb[1 * N_ * N_] = a1;
  bb[2 * N_ * N_] = a2;
  bb[3 * N_ * N_] = a3;
}

// transpose [t][i][j][c] -> [t][j][i][c]
__global__ __launch_bounds__(256) void k_transpose(const float* __restrict__ in,
                                                   float* __restrict__ out) {
  size_t total = (size_t)T_ * N_ * N_ * C_;
  size_t idx = blockIdx.x * (size_t)blockDim.x + threadIdx.x;
  size_t s = (size_t)gridDim.x * blockDim.x;
  for (; idx < total; idx += s) {
    int c = (int)(idx & 63);
    size_t r = idx >> 6;
    int j = (int)(r % N_);
    size_t r2 = r / N_;
    int i = (int)(r2 % N_);
    int t = (int)(r2 / N_);
    out[(((size_t)t * N_ + j) * N_ + i) * 64 + c] = in[idx];
  }
}

// ---------------- WMMA GEMM kernels ------------------------------------------

// qkvg: out{q,k,v,g}[row,64] from zn[row,64] @ {wq,wk,wv,wg}. q scaled 1/4,
// g = sigmoid(. + bg). One 16x16 tile per wave.
__global__ __launch_bounds__(256) void k_qkvg(const bf16* __restrict__ zn,
                                              const float* __restrict__ wq,
                                              const float* __restrict__ wk,
                                              const float* __restrict__ wv,
                                              const float* __restrict__ wg,
                                              const float* __restrict__ bg,
                                              bf16* __restrict__ q, bf16* __restrict__ k,
                                              bf16* __restrict__ v, bf16* __restrict__ g,
                                              int M) {
  int wave = (blockIdx.x * blockDim.x + threadIdx.x) >> 5;
  int lane = threadIdx.x & 31;
  if (wave >= (M / 16) * 16) return;
  int mt = wave >> 4, nt = wave & 15;
  const float* W = (nt < 4) ? wq : (nt < 8) ? wk : (nt < 12) ? wv : wg;
  int nco = (nt & 3) * 16;
  v8f acc = {0, 0, 0, 0, 0, 0, 0, 0};
#pragma unroll
  for (int k0 = 0; k0 < 64; k0 += 32) {
    v16bf a = fragA(lane, [&](int m, int kk) -> bf16 {
      return zn[(size_t)(mt * 16 + m) * 64 + k0 + kk];
    });
    v16bf bw = fragB(lane, [&](int kk, int n) -> bf16 {
      return (bf16)W[(size_t)(k0 + kk) * 64 + nco + n];
    });
    acc = wmma_bf16(a, bw, acc);
  }
  int buf = nt >> 2;
  bf16* out = (buf == 0) ? q : (buf == 1) ? k : (buf == 2) ? v : g;
  storeD(lane, acc, [&](int m, int n, float val) {
    int c = nco + n;
    if (buf == 0) val *= 0.25f;                      // 1/sqrt(C_ATT)
    else if (buf == 3) val = sigmoidf_(val + bg[c]); // gate
    out[(size_t)(mt * 16 + m) * 64 + c] = (bf16)val;
  });
}

// Fused per-(t, i, 16-row q-tile) attention: 4 waves = 4 heads.
// S = QK^T + mask_bias + tri_bias ; softmax ; O = P V ; gated ; out proj + residual.
__global__ __launch_bounds__(128) void k_att(const bf16* __restrict__ q,
                                             const bf16* __restrict__ k,
                                             const bf16* __restrict__ v,
                                             const bf16* __restrict__ g,
                                             const float* __restrict__ tribias,
                                             const float* __restrict__ mask,
                                             const float* __restrict__ wo,
                                             const float* __restrict__ bo,
                                             float* __restrict__ zacc, int maskT) {
  extern __shared__ char smem[];
  float* S  = (float*)smem;                                  // 4 * 16*256 f32
  bf16*  P  = (bf16*)(smem + 4 * 16 * 256 * 4);              // 4 * 16*256 bf16
  float* GO = (float*)(smem + 4 * 16 * 256 * 4 + 4 * 16 * 256 * 2); // 16*64 f32

  int h = threadIdx.x >> 5;
  int lane = threadIdx.x & 31;
  const int qtiles = N_ / 16;
  int qt = blockIdx.x % qtiles;
  int i  = (blockIdx.x / qtiles) % N_;
  int t  = blockIdx.x / (qtiles * N_);
  size_t rowbase = ((size_t)t * N_ + i) * N_;
  float* Sh = S + h * 16 * 256;
  bf16*  Ph = P + h * 16 * 256;

  // S = Q K^T (head dim 16, K padded to 32) + biases
  v16bf aq = fragA(lane, [&](int m, int kk) -> bf16 {
    return (kk < 16) ? q[(rowbase + qt * 16 + m) * 64 + h * 16 + kk] : (bf16)0.f;
  });
  v8f zero = {0, 0, 0, 0, 0, 0, 0, 0};
#pragma unroll 4
  for (int nt2 = 0; nt2 < 16; ++nt2) {
    v16bf bk = fragB(lane, [&](int kk, int n) -> bf16 {
      return (kk < 16) ? k[(rowbase + nt2 * 16 + n) * 64 + h * 16 + kk] : (bf16)0.f;
    });
    v8f acc = wmma_bf16(aq, bk, zero);
    storeD(lane, acc, [&](int m, int n, float val) {
      int kn = nt2 * 16 + n;
      float mval = maskT ? mask[((size_t)t * N_ + kn) * N_ + i] : mask[rowbase + kn];
      val += INFV * (mval - 1.f);
      val += tribias[(((size_t)t * H_ + h) * N_ + qt * 16 + m) * N_ + kn];
      Sh[m * 256 + kn] = val;
    });
  }
  __syncthreads();

  // softmax over 256 keys; lane pair (r, r+16) covers one row's two halves
  {
    int r = lane & 15, seg = lane >> 4;
    float* row = Sh + r * 256;
    float mx = -3.4e38f;
    for (int c2 = seg * 128; c2 < seg * 128 + 128; ++c2) mx = fmaxf(mx, row[c2]);
    mx = fmaxf(mx, __shfl_xor(mx, 16, 32));
    float sum = 0.f;
    for (int c2 = seg * 128; c2 < seg * 128 + 128; ++c2) {
      float e = __expf(row[c2] - mx);
      sum += e;
      Ph[r * 256 + c2] = (bf16)e;
    }
    sum += __shfl_xor(sum, 16, 32);
    float inv = 1.f / sum;
    for (int c2 = seg * 128; c2 < seg * 128 + 128; ++c2)
      Ph[r * 256 + c2] = (bf16)((float)Ph[r * 256 + c2] * inv);
  }
  __syncthreads();

  // O = P @ V (K=256), gated into GO[16][64]
  v8f oacc = {0, 0, 0, 0, 0, 0, 0, 0};
#pragma unroll 2
  for (int kt = 0; kt < 8; ++kt) {
    v16bf ap = fragA(lane, [&](int m, int kk) -> bf16 {
      return Ph[m * 256 + kt * 32 + kk];
    });
    v16bf bv = fragB(lane, [&](int kk, int n) -> bf16 {
      return v[(rowbase + kt * 32 + kk) * 64 + h * 16 + n];
    });
    oacc = wmma_bf16(ap, bv, oacc);
  }
  storeD(lane, oacc, [&](int m, int n, float val) {
    float gv = (float)g[(rowbase + qt * 16 + m) * 64 + h * 16 + n];
    GO[m * 64 + h * 16 + n] = val * gv;
  });
  __syncthreads();

  // out = GO @ wo + bo, accumulate residual into z; wave h owns cols h*16..h*16+15
  v8f facc = {0, 0, 0, 0, 0, 0, 0, 0};
#pragma unroll
  for (int k0 = 0; k0 < 64; k0 += 32) {
    v16bf a2 = fragA(lane, [&](int m, int kk) -> bf16 {
      return (bf16)GO[m * 64 + k0 + kk];
    });
    v16bf b2 = fragB(lane, [&](int kk, int n) -> bf16 {
      return (bf16)wo[(size_t)(k0 + kk) * 64 + h * 16 + n];
    });
    facc = wmma_bf16(a2, b2, facc);
  }
  storeD(lane, facc, [&](int m, int n, float val) {
    size_t o = (rowbase + qt * 16 + m) * 64 + h * 16 + n;
    zacc[o] += val + bo[h * 16 + n];
  });
}

// Tri-mul gated projection: out = mask * sigmoid(zn@Wg + bg) * (zn@Wp + bp),
// written channel-major [t][c][r1][r2] (or [t][c][r2][r1] if swap, for incoming).
__global__ __launch_bounds__(256) void k_gproj(const bf16* __restrict__ zn,
                                               const float* __restrict__ Wg,
                                               const float* __restrict__ Bg,
                                               const float* __restrict__ Wp,
                                               const float* __restrict__ Bp,
                                               const float* __restrict__ mask,
                                               bf16* __restrict__ outT, int swap) {
  int wave = (blockIdx.x * blockDim.x + threadIdx.x) >> 5;
  int lane = threadIdx.x & 31;
  const int M = T_ * N_ * N_;
  if (wave >= (M / 16) * 4) return;
  int mt = wave >> 2, nt = wave & 3;
  v8f accg = {0, 0, 0, 0, 0, 0, 0, 0}, accp = {0, 0, 0, 0, 0, 0, 0, 0};
#pragma unroll
  for (int k0 = 0; k0 < 64; k0 += 32) {
    v16bf a = fragA(lane, [&](int m, int kk) -> bf16 {
      return zn[(size_t)(mt * 16 + m) * 64 + k0 + kk];
    });
    v16bf bg_ = fragB(lane, [&](int kk, int n) -> bf16 {
      return (bf16)Wg[(size_t)(k0 + kk) * 64 + nt * 16 + n];
    });
    v16bf bp_ = fragB(lane, [&](int kk, int n) -> bf16 {
      return (bf16)Wp[(size_t)(k0 + kk) * 64 + nt * 16 + n];
    });
    accg = wmma_bf16(a, bg_, accg);
    accp = wmma_bf16(a, bp_, accp);
  }
  int n = lane & 15, half = (lane >> 4) & 1;
#pragma unroll
  for (int r = 0; r < 8; ++r) {
    int m = r + half * 8;
    int rg = mt * 16 + m;
    int c = nt * 16 + n;
    int t = rg / (N_ * N_), rr = rg % (N_ * N_);
    int r1 = rr / N_, r2 = rr % N_;
    float val = mask[rg] * sigmoidf_(accg[r] + Bg[c]) * (accp[r] + Bp[c]);
    size_t o = swap ? ((((size_t)t * 64 + c) * N_ + r2) * N_ + r1)
                    : ((((size_t)t * 64 + c) * N_ + r1) * N_ + r2);
    outT[o] = (bf16)val;
  }
}

// Batched per-channel GEMM: X[mat][i][j] = sum_k P[mat][i][k] * Q[mat][j][k]
__global__ __launch_bounds__(256) void k_chan_gemm(const bf16* __restrict__ P,
                                                   const bf16* __restrict__ Q,
                                                   float* __restrict__ X) {
  int wave = (blockIdx.x * blockDim.x + threadIdx.x) >> 5;
  int lane = threadIdx.x & 31;
  const int tilesPerMat = (N_ / 16) * (N_ / 16);
  if (wave >= T_ * 64 * tilesPerMat) return;
  int mat = wave / tilesPerMat;
  int tl = wave % tilesPerMat;
  int it = tl >> 4, jt = tl & 15;
  const bf16* Pb = P + (size_t)mat * N_ * N_;
  const bf16* Qb = Q + (size_t)mat * N_ * N_;
  v8f acc = {0, 0, 0, 0, 0, 0, 0, 0};
#pragma unroll 2
  for (int kt = 0; kt < 8; ++kt) {
    v16bf a = fragA(lane, [&](int m, int kk) -> bf16 {
      return Pb[(size_t)(it * 16 + m) * N_ + kt * 32 + kk];
    });
    v16bf b = fragB(lane, [&](int kk, int n) -> bf16 {
      return Qb[(size_t)(jt * 16 + n) * N_ + kt * 32 + kk];
    });
    acc = wmma_bf16(a, b, acc);
  }
  float* Xb = X + (size_t)mat * N_ * N_;
  storeD(lane, acc, [&](int m, int n, float val) {
    Xb[(size_t)(it * 16 + m) * N_ + jt * 16 + n] = val;
  });
}

// Generic GEMM: out_bf16[M,Nc] = act(A[M,K] @ W[K,Nc] + bias). act: 0 none/1 sigmoid/2 relu
__global__ __launch_bounds__(256) void k_gemm_act(const bf16* __restrict__ A, int K,
                                                  const float* __restrict__ W, int Nc,
                                                  const float* __restrict__ bias,
                                                  bf16* __restrict__ out, int M, int act) {
  int wave = (blockIdx.x * blockDim.x + threadIdx.x) >> 5;
  int lane = threadIdx.x & 31;
  int ntiles = Nc / 16;
  if (wave >= (M / 16) * ntiles) return;
  int mt = wave / ntiles, nt = wave % ntiles;
  v8f acc = {0, 0, 0, 0, 0, 0, 0, 0};
  for (int k0 = 0; k0 < K; k0 += 32) {
    v16bf a = fragA(lane, [&](int m, int kk) -> bf16 {
      return A[(size_t)(mt * 16 + m) * K + k0 + kk];
    });
    v16bf b = fragB(lane, [&](int kk, int n) -> bf16 {
      return (bf16)W[(size_t)(k0 + kk) * Nc + nt * 16 + n];
    });
    acc = wmma_bf16(a, b, acc);
  }
  storeD(lane, acc, [&](int m, int n, float val) {
    int c = nt * 16 + n;
    val += bias[c];
    if (act == 1) val = sigmoidf_(val);
    else if (act == 2) val = fmaxf(val, 0.f);
    out[(size_t)(mt * 16 + m) * Nc + c] = (bf16)val;
  });
}

// Residual GEMM: zacc[M,64] += (A[M,K] @ W[K,64] + bias) [* gate]
__global__ __launch_bounds__(256) void k_gemm_accum(const bf16* __restrict__ A, int K,
                                                    const float* __restrict__ W,
                                                    const float* __restrict__ bias,
                                                    const bf16* __restrict__ gate,
                                                    float* __restrict__ zacc, int M) {
  int wave = (blockIdx.x * blockDim.x + threadIdx.x) >> 5;
  int lane = threadIdx.x & 31;
  if (wave >= (M / 16) * 4) return;
  int mt = wave >> 2, nt = wave & 3;
  v8f acc = {0, 0, 0, 0, 0, 0, 0, 0};
  for (int k0 = 0; k0 < K; k0 += 32) {
    v16bf a = fragA(lane, [&](int m, int kk) -> bf16 {
      return A[(size_t)(mt * 16 + m) * K + k0 + kk];
    });
    v16bf b = fragB(lane, [&](int kk, int n) -> bf16 {
      return (bf16)W[(size_t)(k0 + kk) * 64 + nt * 16 + n];
    });
    acc = wmma_bf16(a, b, acc);
  }
  storeD(lane, acc, [&](int m, int n, float val) {
    int c = nt * 16 + n;
    float o = val + bias[c];
    size_t off = (size_t)(mt * 16 + m) * 64 + c;
    if (gate) o *= (float)gate[off];
    zacc[off] += o;
  });
}

// ---------------- host orchestration ----------------------------------------

extern "C" void kernel_launch(void* const* d_in, const int* in_sizes, int n_in,
                              void* d_out, int out_size, void* d_ws, size_t ws_size,
                              hipStream_t stream) {
  (void)in_sizes; (void)n_in; (void)out_size; (void)ws_size;
  const float* z    = (const float*)d_in[0];
  const float* mask = (const float*)d_in[1];
  float* zbuf = (float*)d_out;

  // d_in layout: 2..11 att_start, 12..21 att_end, 22..37 mul_out, 38..53 mul_in,
  // 54..59 trans (flattened in setup_inputs dict order).
  const float* a0[10]; const float* a1[10];
  for (int i = 0; i < 10; ++i) { a0[i] = (const float*)d_in[2 + i];
                                 a1[i] = (const float*)d_in[12 + i]; }
  const float* m0[16]; const float* m1[16];
  for (int i = 0; i < 16; ++i) { m0[i] = (const float*)d_in[22 + i];
                                 m1[i] = (const float*)d_in[38 + i]; }
  const float* tr[6];
  for (int i = 0; i < 6; ++i) tr[i] = (const float*)d_in[54 + i];

  // workspace carve-out (~114 MB)
  char* w = (char*)d_ws;
  const size_t SB = (size_t)T_ * N_ * N_ * 64 * 2;  // 16 MiB bf16 buffer
  bf16* zn = (bf16*)w;
  bf16* B1 = (bf16*)(w + 1 * SB);
  bf16* B2 = (bf16*)(w + 2 * SB);   // B1+B2 contiguous -> 128-wide transition buffer
  bf16* B3 = (bf16*)(w + 3 * SB);
  bf16* B4 = (bf16*)(w + 4 * SB);
  float* E  = (float*)(w + 5 * SB); // 32 MiB f32 (transpose buffer / channel-major X)
  float* tb = (float*)(w + 7 * SB); // tri-bias [T,H,N,N] f32

  const int R = T_ * N_ * N_;                 // 131072 rows
  const size_t TOT = (size_t)R * 64;
  const size_t attShm = 4 * 16 * 256 * 4 + 4 * 16 * 256 * 2 + 16 * 64 * 4; // 100 KiB

  k_copy<<<4096, 256, 0, stream>>>(z, zbuf, TOT);

  // ---- triangle attention (starting node) on zbuf ----
  k_ln<<<R / 8, 256, 0, stream>>>(zbuf, a0[0], a0[1], zn, R);
  k_tribias<<<(R + 255) / 256, 256, 0, stream>>>(zn, a0[5], tb, R);
  k_qkvg<<<16384, 256, 0, stream>>>(zn, a0[2], a0[3], a0[4], a0[6], a0[7],
                                    B1, B2, B3, B4, R);
  k_att<<<T_ * N_ * (N_ / 16), 128, attShm, stream>>>(B1, B2, B3, B4, tb, mask,
                                                      a0[8], a0[9], zbuf, 0);

  // ---- triangle attention (ending node): work on transpose ----
  k_transpose<<<4096, 256, 0, stream>>>(zbuf, E);
  k_ln<<<R / 8, 256, 0, stream>>>(E, a1[0], a1[1], zn, R);
  k_tribias<<<(R + 255) / 256, 256, 0, stream>>>(zn, a1[5], tb, R);
  k_qkvg<<<16384, 256, 0, stream>>>(zn, a1[2], a1[3], a1[4], a1[6], a1[7],
                                    B1, B2, B3, B4, R);
  k_att<<<T_ * N_ * (N_ / 16), 128, attShm, stream>>>(B1, B2, B3, B4, tb, mask,
                                                      a1[8], a1[9], E, 1);
  k_transpose<<<4096, 256, 0, stream>>>(E, zbuf);

  // ---- triangle multiplication (outgoing, then incoming) ----
  for (int pass = 0; pass < 2; ++pass) {
    const float** m = pass == 0 ? m0 : m1;
    int swap = pass;  // 0: outgoing (ik,jk); 1: incoming (ki,kj) via transposed write
    k_ln<<<R / 8, 256, 0, stream>>>(zbuf, m[0], m[1], zn, R);
    k_gproj<<<4096, 256, 0, stream>>>(zn, m[2], m[3], m[4], m[5], mask, B1, swap);
    k_gproj<<<4096, 256, 0, stream>>>(zn, m[6], m[7], m[8], m[9], mask, B2, swap);
    k_chan_gemm<<<4096, 256, 0, stream>>>(B1, B2, E);
    k_ln_cm<<<R / 8, 256, 0, stream>>>(E, m[10], m[11], B3);
    k_gemm_act<<<4096, 256, 0, stream>>>(zn, 64, m[14], 64, m[15], B4, R, 1);
    k_gemm_accum<<<4096, 256, 0, stream>>>(B3, 64, m[12], m[13], B4, zbuf, R);
  }

  // ---- transition ----
  k_ln<<<R / 8, 256, 0, stream>>>(zbuf, tr[0], tr[1], zn, R);
  k_gemm_act<<<8192, 256, 0, stream>>>(zn, 64, tr[2], 128, tr[3], B1, R, 2);
  k_gemm_accum<<<4096, 256, 0, stream>>>(B1, 128, tr[4], tr[5], nullptr, zbuf, R);
}
